// RealSHT_25933012533836
// MI455X (gfx1250) — matrix operand order repaired and checked by reading, compiled, not run on hardware
//
#include <hip/hip_runtime.h>
#include <hip/hip_bf16.h>

// ---------------------------------------------------------------------------
// RealSHT for MI455X (gfx1250, wave32, WMMA + async-to-LDS).
// Stage 1: bf16 hi/lo DFT factor planes F_hi/F_lo[col=2m+p][n] (~2.2 MB, L2)
// Stage 2: DFT as GEMM  xm[bck,2m+p] = x[bck,n] * F[n,2m+p]  (split-bf16 WMMA)
// Stage 3: Legendre     out[bc,l,m]  = xm[m][bc,p][k] * w[m][l][k]
// Split precision: v = hi(bf16) + lo(bf16 residual); D += AhBh + AhBl + AlBh.
// hi/lo live in SEPARATE planes everywhere -> WMMA fragments are contiguous
// bf16 runs (pure ds_load_b128) and pure-copy stagings use
// global_load_async_to_lds_b64/b128 (ASYNCcnt, no VGPR round-trip).
// ---------------------------------------------------------------------------

typedef __attribute__((ext_vector_type(16))) __bf16          v16bf;
typedef __attribute__((ext_vector_type(8)))  unsigned short  v8us;
typedef __attribute__((ext_vector_type(8)))  float           v8f;

#define NLAT   360
#define NLON   720
#define LMAX   360
#define MMAX   361
#define BC     256        // B*C = 8*32
#define NCOL   722        // 2*MMAX (re/im interleaved as GEMM columns)
#define NPAD   736        // 23*32 : padded N and K for stage 2
#define KB     384        // 12*32 : padded K for stage 3
#define PITCH  40         // LDS halfword pitch: 16B-aligned rows, conflict-free

// LDS byte offset of a __shared__ pointer (addrspace(3) ptrtoint)
#define LDS_OFF(p) ((unsigned)(size_t)(__attribute__((address_space(3))) const void*)(p))

// async global->LDS copies (tracked by ASYNCcnt)
static __device__ __forceinline__ void async_copy_b64(unsigned lds_off,
                                                      const void* gptr) {
    asm volatile("global_load_async_to_lds_b64 %0, %1, off"
                 :: "v"(lds_off), "v"(gptr) : "memory");
}
static __device__ __forceinline__ void async_copy_b128x2(unsigned lds_off,
                                                         const void* gptr) {
    asm volatile("global_load_async_to_lds_b128 %0, %1, off"
                 :: "v"(lds_off), "v"(gptr) : "memory");
    asm volatile("global_load_async_to_lds_b128 %0, %1, off offset:16"
                 :: "v"(lds_off), "v"(gptr) : "memory");
}
static __device__ __forceinline__ void wait_async() {
    asm volatile("s_wait_asynccnt 0x0" ::: "memory");
}

// fp32 -> bf16 hi + bf16 residual
static __device__ __forceinline__ void split1(float v, unsigned short& h,
                                              unsigned short& l) {
    __bf16 hb = (__bf16)v;
    __bf16 lb = (__bf16)(v - (float)hb);
    h = __builtin_bit_cast(unsigned short, hb);
    l = __builtin_bit_cast(unsigned short, lb);
}
// two fp32 -> packed u32 per plane (two adjacent k slots)
static __device__ __forceinline__ void split2(float v0, float v1,
                                              unsigned& hi, unsigned& lo) {
    unsigned short h0, l0, h1, l1;
    split1(v0, h0, l0);
    split1(v1, h1, l1);
    hi = (unsigned)h0 | ((unsigned)h1 << 16);
    lo = (unsigned)l0 | ((unsigned)l1 << 16);
}

// A fragment: elements 0..7 at base[0..7], 8..15 at base[16..23] (ISA layout)
static __device__ __forceinline__ v16bf frag_a(const unsigned short* base) {
    v8us x0 = *(const v8us*)(base);
    v8us x1 = *(const v8us*)(base + 16);
    return __builtin_bit_cast(v16bf, __builtin_shufflevector(
        x0, x1, 0, 1, 2, 3, 4, 5, 6, 7, 8, 9, 10, 11, 12, 13, 14, 15));
}
// B fragment: elements 0..7 at base[0..7], 8..15 at base[16..23] (ISA layout)
static __device__ __forceinline__ v16bf frag_b(const unsigned short* base) {
    v8us x0 = *(const v8us*)(base);
    v8us x1 = *(const v8us*)(base + 16);
    return __builtin_bit_cast(v16bf, __builtin_shufflevector(
        x0, x1, 0, 1, 2, 3, 4, 5, 6, 7, 8, 9, 10, 11, 12, 13, 14, 15));
}

// ---------------------------------------------------------------------------
// Kernel 1: DFT factor planes. col = 2m+p (p=0: cos, p=1: -sin), scaled by
// 2*pi/NLON, exact integer phase (m*n)%720. Pads zero-filled.
// ---------------------------------------------------------------------------
__global__ void sht_build_dft(unsigned short* __restrict__ Fh,
                              unsigned short* __restrict__ Fl) {
    int idx = blockIdx.x * blockDim.x + threadIdx.x;
    if (idx >= NPAD * NPAD) return;
    int col = idx / NPAD;
    int n   = idx % NPAD;
    float v = 0.0f;
    if (col < NCOL && n < NLON) {
        int m = col >> 1;
        int t = (m * n) % NLON;                       // exact phase index
        const float scale = 6.2831853071795864769f / (float)NLON;
        float th = (float)t * scale;
        float s, c;
        sincosf(th, &s, &c);
        v = (col & 1) ? (-scale * s) : (scale * c);
    }
    unsigned short h, l;
    split1(v, h, l);
    Fh[idx] = h;
    Fl[idx] = l;
}

// ---------------------------------------------------------------------------
// Kernel 2: DFT GEMM. Grid: x = N-block (23 of 32 cols), y = row-block (720
// of 128 rows). 8 waves/WG, each wave owns one 16-row tile x two 16-col tiles.
// Output: hi/lo ushort planes xmh/xml, layout [m][bc][p][KB].
// ---------------------------------------------------------------------------
__global__ void __launch_bounds__(256, 2)
sht_dft_gemm(const float* __restrict__ x, const unsigned short* __restrict__ Fh,
             const unsigned short* __restrict__ Fl,
             unsigned short* __restrict__ xmh, unsigned short* __restrict__ xml) {
    __shared__ __align__(16) unsigned short ldsAh[128 * PITCH];
    __shared__ __align__(16) unsigned short ldsAl[128 * PITCH];
    __shared__ __align__(16) unsigned short ldsBh[32 * PITCH];
    __shared__ __align__(16) unsigned short ldsBl[32 * PITCH];

    const int tid  = threadIdx.x;
    const int wave = tid >> 5;
    const int lane = tid & 31;
    const int colBase = blockIdx.x * 32;
    const int rowBase = blockIdx.y * 128;

    // per-thread staging geometry (invariant over K)
    const int rloc = tid >> 1;
    const int half = (tid & 1) * 16;
    const int rg = rowBase + rloc;
    const int bcS = rg / NLAT;
    const int klS = rg - bcS * NLAT;
    const float* rowp = x + ((size_t)(bcS * NLAT + klS)) * NLON;
    const int cl = tid >> 3;               // F-tile: col 0..31
    const int q  = (tid & 7) * 4;          // F-tile: halfword quad 0..28
    const unsigned ldsBoH = LDS_OFF(&ldsBh[cl * PITCH + q]);
    const unsigned ldsBoL = LDS_OFF(&ldsBl[cl * PITCH + q]);

    v8f acc0 = {}; v8f acc1 = {};

    for (int ks = 0; ks < NPAD; ks += 32) {
        __syncthreads();
        // F tile: pure async copy into LDS planes (8B per thread per plane)
        async_copy_b64(ldsBoH, Fh + (size_t)(colBase + cl) * NPAD + ks + q);
        async_copy_b64(ldsBoL, Fl + (size_t)(colBase + cl) * NPAD + ks + q);
        // x tile: fp32 -> split planes. 720 % 16 == 0 -> whole 16-run uniform.
        if (ks + half < NLON) {
            const float4* s4 = (const float4*)(rowp + ks + half);
            if (ks + 32 < NLON) __builtin_prefetch(s4 + 8, 0, 3);
            #pragma unroll
            for (int qq = 0; qq < 4; ++qq) {
                float4 f = s4[qq];
                unsigned h01, l01, h23, l23;
                split2(f.x, f.y, h01, l01);
                split2(f.z, f.w, h23, l23);
                *(uint2*)&ldsAh[rloc * PITCH + half + qq * 4] = make_uint2(h01, h23);
                *(uint2*)&ldsAl[rloc * PITCH + half + qq * 4] = make_uint2(l01, l23);
            }
        } else {
            #pragma unroll
            for (int qq = 0; qq < 4; ++qq) {
                *(uint2*)&ldsAh[rloc * PITCH + half + qq * 4] = make_uint2(0u, 0u);
                *(uint2*)&ldsAl[rloc * PITCH + half + qq * 4] = make_uint2(0u, 0u);
            }
        }
        wait_async();
        __syncthreads();

        const int mr = lane & 15, g = lane >> 4;
        v16bf a_hi = frag_a(&ldsAh[(wave * 16 + mr) * PITCH + g * 8]);
        v16bf a_lo = frag_a(&ldsAl[(wave * 16 + mr) * PITCH + g * 8]);

        #pragma unroll
        for (int ct = 0; ct < 2; ++ct) {
            int co = (ct * 16 + (lane & 15)) * PITCH + g * 8;
            v16bf b_hi = frag_b(&ldsBh[co]);
            v16bf b_lo = frag_b(&ldsBl[co]);
            v8f& acc = ct ? acc1 : acc0;
            acc = __builtin_amdgcn_wmma_f32_16x16x32_bf16(false, a_hi, false, b_hi, (short)0, acc, false, false);
            acc = __builtin_amdgcn_wmma_f32_16x16x32_bf16(false, a_hi, false, b_lo, (short)0, acc, false, false);
            acc = __builtin_amdgcn_wmma_f32_16x16x32_bf16(false, a_lo, false, b_hi, (short)0, acc, false, false);
        }
    }

    // epilogue: lane's 8 accum values sit on 8 consecutive k of one (bc,p,m)
    // row (8-aligned runs never straddle the 360 boundary) -> one uint4/plane.
    {
        int nc = lane & 15, g = lane >> 4;
        int gr0 = rowBase + wave * 16 + g * 8;
        int bc = gr0 / NLAT;
        int kl0 = gr0 - bc * NLAT;
        #pragma unroll
        for (int ct = 0; ct < 2; ++ct) {
            int gc = colBase + ct * 16 + nc;
            if (gc < NCOL) {
                int msh = gc >> 1, p = gc & 1;
                const v8f& acc = ct ? acc1 : acc0;
                unsigned hh[4], ll[4];
                #pragma unroll
                for (int qq = 0; qq < 4; ++qq)
                    split2(acc[2 * qq], acc[2 * qq + 1], hh[qq], ll[qq]);
                size_t base = (((size_t)msh * BC + bc) * 2 + p) * KB + kl0;
                *(uint4*)&xmh[base] = make_uint4(hh[0], hh[1], hh[2], hh[3]);
                *(uint4*)&xml[base] = make_uint4(ll[0], ll[1], ll[2], ll[3]);
            }
        }
    }
}

// ---------------------------------------------------------------------------
// Kernel 3: Legendre GEMMs. Grid: x = l-tile (23), y = row-block (4 of 128
// rows, row = bc*2+p), z = m-block (46 of 8). 8 waves, 1 tile each; loops over
// 8 m values; K loop 12x32 (weights zero-filled past 360 -> pad-safe; xm pad
// garbage is annihilated by the zero weights side).
// ---------------------------------------------------------------------------
__global__ void __launch_bounds__(256, 2)
sht_leg_gemm(const unsigned short* __restrict__ xmh,
             const unsigned short* __restrict__ xml,
             const float* __restrict__ w, float* __restrict__ out) {
    __shared__ __align__(16) unsigned short ldsAh[128 * PITCH];
    __shared__ __align__(16) unsigned short ldsAl[128 * PITCH];
    __shared__ __align__(16) unsigned short ldsBh[16 * PITCH];
    __shared__ __align__(16) unsigned short ldsBl[16 * PITCH];
    __shared__ float stage[64 * 289];   // [bcl]*289 + [l]*18 + [2*ml+p]

    const int tid  = threadIdx.x;
    const int wave = tid >> 5;
    const int lane = tid & 31;
    const int lBase   = blockIdx.x * 16;
    const int rowBase = blockIdx.y * 128;
    const int mBase   = blockIdx.z * 8;

    const int rloc = tid >> 1;
    const int half = (tid & 1) * 16;
    const unsigned ldsAoH = LDS_OFF(&ldsAh[rloc * PITCH + half]);
    const unsigned ldsAoL = LDS_OFF(&ldsAl[rloc * PITCH + half]);
    const int ll = tid >> 4;          // weights tile: l row 0..15
    const int k0 = (tid & 15) * 2;    // weights tile: k pair 0..30

    for (int ml = 0; ml < 8; ++ml) {
        const int m = mBase + ml;
        const bool mvalid = (m < MMAX);       // uniform across the workgroup
        v8f acc = {};
        for (int ks = 0; ks < KB && mvalid; ks += 32) {
            __syncthreads();
            // A tile: pure async copies from the xm planes (32B/thread/plane)
            size_t gbase = ((size_t)m * (BC * 2) + rowBase + rloc) * KB + ks + half;
            async_copy_b128x2(ldsAoH, xmh + gbase);
            async_copy_b128x2(ldsAoL, xml + gbase);
            // B tile: weights fp32 -> split planes, transposed, zero pad.
            {
                int lg = lBase + ll;
                int kg = ks + k0;
                float2 v = make_float2(0.0f, 0.0f);
                if (lg < LMAX && kg < NLAT)   // kg+1 valid iff kg valid (360 even)
                    v = *(const float2*)&w[((size_t)m * LMAX + lg) * NLAT + kg];
                unsigned hi, lo;
                split2(v.x, v.y, hi, lo);
                *(unsigned*)&ldsBh[ll * PITCH + k0] = hi;
                *(unsigned*)&ldsBl[ll * PITCH + k0] = lo;
            }
            wait_async();
            __syncthreads();

            const int mr = lane & 15, g = lane >> 4;
            v16bf a_hi = frag_a(&ldsAh[(wave * 16 + mr) * PITCH + g * 8]);
            v16bf a_lo = frag_a(&ldsAl[(wave * 16 + mr) * PITCH + g * 8]);
            int co = (lane & 15) * PITCH + g * 8;
            v16bf b_hi = frag_b(&ldsBh[co]);
            v16bf b_lo = frag_b(&ldsBl[co]);

            acc = __builtin_amdgcn_wmma_f32_16x16x32_bf16(false, a_hi, false, b_hi, (short)0, acc, false, false);
            acc = __builtin_amdgcn_wmma_f32_16x16x32_bf16(false, a_hi, false, b_lo, (short)0, acc, false, false);
            acc = __builtin_amdgcn_wmma_f32_16x16x32_bf16(false, a_lo, false, b_hi, (short)0, acc, false, false);
        }
        if (mvalid) {   // scatter fragment into transpose staging buffer
            int nc = lane & 15, g = lane >> 4;
            #pragma unroll
            for (int j = 0; j < 8; ++j) {
                int rl = wave * 16 + g * 8 + j;
                stage[(rl >> 1) * 289 + nc * 18 + ml * 2 + (rl & 1)] = acc[j];
            }
        }
    }
    __syncthreads();

    // Coalesced store: 1024 (bcl,l) runs; each run = 16 contiguous floats
    // (8 m x {re,im}) in out[bc][l][m][p].
    const bool fullM = (mBase + 8 <= MMAX);   // uniform
    #pragma unroll
    for (int rr = 0; rr < 4; ++rr) {
        int run = rr * 256 + tid;
        int bcl = run >> 4;
        int nc  = run & 15;
        int lg  = lBase + nc;
        if (lg >= LMAX) continue;
        int bc = (rowBase >> 1) + bcl;
        size_t base = (((size_t)bc * LMAX + lg) * MMAX + mBase) * 2;
        const float* s = &stage[bcl * 289 + nc * 18];
        if (fullM) {
            #pragma unroll
            for (int i = 0; i < 8; ++i)
                *(float2*)&out[base + i * 2] = *(const float2*)&s[i * 2];
        } else {
            #pragma unroll
            for (int i = 0; i < 8; ++i)
                if (mBase + i < MMAX)
                    *(float2*)&out[base + i * 2] = *(const float2*)&s[i * 2];
        }
    }
}

// ---------------------------------------------------------------------------
extern "C" void kernel_launch(void* const* d_in, const int* in_sizes, int n_in,
                              void* d_out, int out_size, void* d_ws, size_t ws_size,
                              hipStream_t stream) {
    const float* x = (const float*)d_in[0];      // (8,32,360,720) f32
    const float* w = (const float*)d_in[1];      // (361,360,360) f32
    float* out = (float*)d_out;                  // complex64 interleaved

    // Workspace: F_hi [0, 1.04MB); F_lo [1.0625MB); xm hi plane at 4MB,
    // xm lo plane at 148MB: ushort [361][256][2][384] each (~135 MB/plane).
    unsigned short* Fh  = (unsigned short*)d_ws;
    unsigned short* Fl  = (unsigned short*)((char*)d_ws + (size_t)0x110000);
    unsigned short* xmh = (unsigned short*)((char*)d_ws + ((size_t)4 << 20));
    unsigned short* xml = (unsigned short*)((char*)d_ws + ((size_t)148 << 20));

    sht_build_dft<<<(NPAD * NPAD + 255) / 256, 256, 0, stream>>>(Fh, Fl);
    sht_dft_gemm<<<dim3(23, 720), 256, 0, stream>>>(x, Fh, Fl, xmh, xml);
    sht_leg_gemm<<<dim3(23, 4, 46), 256, 0, stream>>>(xmh, xml, w, out);
}
